// Attention_25812753449422
// MI455X (gfx1250) — matrix operand order repaired and checked by reading, not compiled, not run on hardware
//
#include <hip/hip_runtime.h>

#define SEQ  2048
#define BSZ  2
#define HID  2048
#define NH   16
#define HD   128

typedef __attribute__((ext_vector_type(2)))  __bf16 v2bf;
typedef __attribute__((ext_vector_type(8)))  __bf16 v8bf;
typedef __attribute__((ext_vector_type(16))) __bf16 v16bf;
typedef __attribute__((ext_vector_type(8)))  float  v8f;
typedef __attribute__((ext_vector_type(4)))  unsigned int v4u;
typedef __attribute__((ext_vector_type(8)))  unsigned int v8u;

__device__ __forceinline__ unsigned short f2bf(float f) {
    __bf16 h = (__bf16)f;                     // native fptrunc (RNE)
    return __builtin_bit_cast(unsigned short, h);
}
__device__ __forceinline__ unsigned int pk2(float a, float b) {
    v2bf v = { (__bf16)a, (__bf16)b };        // lowers to packed cvt
    return __builtin_bit_cast(unsigned int, v);
}
__device__ __forceinline__ v16bf cat16(v8bf lo, v8bf hi) {
    return __builtin_shufflevector(lo, hi, 0,1,2,3,4,5,6,7,8,9,10,11,12,13,14,15);
}
__device__ __forceinline__ v8f zero8() {
    v8f z = {0.f,0.f,0.f,0.f,0.f,0.f,0.f,0.f};
    return z;
}
__device__ __forceinline__ v8f wmma_bf16(v16bf a, v16bf b, v8f c) {
    return __builtin_amdgcn_wmma_f32_16x16x32_bf16(false, a, false, b, (short)0, c, false, false);
}
// A fragment (16x32, M=lane%16; lanes<16: K 0-7|16-23, lanes>=16: K 8-15|24-31)
__device__ __forceinline__ v16bf load_afrag(const unsigned short* rowPtr, int half) {
    const unsigned short* p = rowPtr + half * 8;
    return cat16(*(const v8bf*)p, *(const v8bf*)(p + 16));
}
// B fragment (32x16, N=lane%16; lanes<16: K 0-7|16-23, lanes>=16: K 8-15|24-31)
__device__ __forceinline__ v16bf load_bfrag(const unsigned short* colPtr, int half) {
    const unsigned short* p = colPtr + half * 8;
    return cat16(*(const v8bf*)p, *(const v8bf*)(p + 16));
}

// ---------------------------------------------------------------------------
// Tensor Data Mover: 2D bf16 tile load Global -> LDS (2-SGPR-group descriptor,
// VADDR2/3 = NULL => up-to-2D tensor).  Issue once per block (wave 0), wait
// with s_wait_tensorcnt.  tile0 = contiguous dim (elements), tile1 = rows,
// stride0 = row stride in elements.  Tensor dims == tile dims (no OOB).
// ---------------------------------------------------------------------------
__device__ __forceinline__ void tdm_load_2d(unsigned lds_addr, const void* gptr,
                                            unsigned tile0, unsigned tile1,
                                            unsigned stride0) {
    unsigned long long ga = (unsigned long long)gptr;
    v4u g0;
    g0[0] = 1u;                                     // count=1, user descriptor
    g0[1] = lds_addr;                               // LDS byte address
    g0[2] = (unsigned)ga;                           // global_addr[31:0]
    g0[3] = (unsigned)(ga >> 32) | (2u << 30);      // global_addr[56:32] | type=2
    v8u g1;
    g1[0] = 1u << 16;                               // data_size = 2 bytes
    g1[1] = (tile0 & 0xffffu) << 16;                // tensor_dim0[15:0]
    g1[2] = ((tile0 >> 16) & 0xffffu) | ((tile1 & 0xffffu) << 16); // td0 hi | td1 lo
    g1[3] = ((tile1 >> 16) & 0xffffu) | ((tile0 & 0xffffu) << 16); // td1 hi | tile_dim0
    g1[4] = tile1 & 0xffffu;                        // tile_dim1 (tile_dim2 = 0)
    g1[5] = stride0;                                // tensor_dim0_stride[31:0]
    g1[6] = 0u;                                     // stride0 hi | stride1 lo
    g1[7] = 0u;
    asm volatile("tensor_load_to_lds %0, %1" :: "s"(g0), "s"(g1) : "memory");
}
__device__ __forceinline__ unsigned lds_off(const void* p) {
    return (unsigned)(unsigned long long)p;         // low 32 bits = LDS offset
}

// ---------------------------------------------------------------------------
// Kernel 1: QKV projection.  C[m,n] = X[m,:]·Wqkv[n,:] + bias[n]
// m = s*BSZ + b (4096 rows), n in [0,6144).  Scatter to Q/K ([b,h,s,d]) and
// V transposed ([b,h,d,s]) in bf16; fold 1/sqrt(HD) into Q.
// (fp32 -> bf16 conversion required, so staging stays on the VALU path.)
// ---------------------------------------------------------------------------
__global__ __launch_bounds__(256) void qkv_gemm_kernel(
    const float* __restrict__ X, const float* __restrict__ Wqkv,
    const float* __restrict__ bias,
    unsigned short* __restrict__ qws, unsigned short* __restrict__ kws,
    unsigned short* __restrict__ vtws)
{
    __shared__ unsigned short lA[128 * 32];
    __shared__ unsigned short lB[128 * 32];

    const int tid  = threadIdx.x;
    const int lane = tid & 31;
    const int w    = tid >> 5;
    const int half = lane >> 4;
    const int r    = lane & 15;

    const int m0 = blockIdx.y * 128;
    const int n0 = blockIdx.x * 128;
    const int wm = (w & 3) * 32;
    const int wn = (w >> 2) * 64;

    v8f acc[2][4];
    #pragma unroll
    for (int i = 0; i < 2; i++)
        #pragma unroll
        for (int j = 0; j < 4; j++) acc[i][j] = zero8();

    const int rowS = tid >> 1;          // 0..127
    const int colS = (tid & 1) * 16;    // 0 or 16

    for (int k0 = 0; k0 < HID; k0 += 32) {
        {   // stage X tile (fp32 -> bf16)
            const float4* s4 = (const float4*)(X + (size_t)(m0 + rowS) * HID + k0 + colS);
            float4 f0 = s4[0], f1 = s4[1], f2 = s4[2], f3 = s4[3];
            uint4 lo, hi;
            lo.x = pk2(f0.x, f0.y); lo.y = pk2(f0.z, f0.w);
            lo.z = pk2(f1.x, f1.y); lo.w = pk2(f1.z, f1.w);
            hi.x = pk2(f2.x, f2.y); hi.y = pk2(f2.z, f2.w);
            hi.z = pk2(f3.x, f3.y); hi.w = pk2(f3.z, f3.w);
            uint4* d4 = (uint4*)&lA[rowS * 32 + colS];
            d4[0] = lo; d4[1] = hi;
        }
        {   // stage W tile (fp32 -> bf16)
            const float4* s4 = (const float4*)(Wqkv + (size_t)(n0 + rowS) * HID + k0 + colS);
            float4 f0 = s4[0], f1 = s4[1], f2 = s4[2], f3 = s4[3];
            uint4 lo, hi;
            lo.x = pk2(f0.x, f0.y); lo.y = pk2(f0.z, f0.w);
            lo.z = pk2(f1.x, f1.y); lo.w = pk2(f1.z, f1.w);
            hi.x = pk2(f2.x, f2.y); hi.y = pk2(f2.z, f2.w);
            hi.z = pk2(f3.x, f3.y); hi.w = pk2(f3.z, f3.w);
            uint4* d4 = (uint4*)&lB[rowS * 32 + colS];
            d4[0] = lo; d4[1] = hi;
        }
        __syncthreads();

        v16bf af[2], bfv[4];
        #pragma unroll
        for (int i = 0; i < 2; i++)
            af[i] = load_afrag(&lA[(wm + i * 16 + r) * 32], half);
        #pragma unroll
        for (int j = 0; j < 4; j++)
            bfv[j] = load_bfrag(&lB[(wn + j * 16 + r) * 32], half);
        #pragma unroll
        for (int i = 0; i < 2; i++)
            #pragma unroll
            for (int j = 0; j < 4; j++)
                acc[i][j] = wmma_bf16(af[i], bfv[j], acc[i][j]);
        __syncthreads();
    }

    const float qscale = 0.08838834764831845f;   // 1/sqrt(128)
    #pragma unroll
    for (int j = 0; j < 4; j++) {
        const int n  = n0 + wn + j * 16 + r;
        const int h  = n / 384;
        const int rc = n % 384;
        const float bv = bias[n];
        #pragma unroll
        for (int i = 0; i < 2; i++) {
            #pragma unroll
            for (int vr = 0; vr < 8; vr++) {
                const int m = m0 + wm + i * 16 + half * 8 + vr;
                const int s  = m >> 1;
                const int bb = m & 1;
                const int bh = bb * NH + h;
                float v = acc[i][j][vr] + bv;
                if (rc < 128) {
                    qws[((size_t)bh * SEQ + s) * HD + rc] = f2bf(v * qscale);
                } else if (rc < 256) {
                    kws[((size_t)bh * SEQ + s) * HD + (rc - 128)] = f2bf(v);
                } else {
                    vtws[((size_t)bh * HD + (rc - 256)) * SEQ + s] = f2bf(v);
                }
            }
        }
    }
}

// ---------------------------------------------------------------------------
// Kernel 2: causal flash attention.  One block per (b*NH, 128 query rows).
// Wave w owns 16 query rows; K/V^T tiles of 32 keys staged in LDS by the TDM.
// ---------------------------------------------------------------------------
__global__ __launch_bounds__(256) void attn_kernel(
    const unsigned short* __restrict__ qws, const unsigned short* __restrict__ kws,
    const unsigned short* __restrict__ vtws, unsigned short* __restrict__ ctx)
{
    __shared__ unsigned short kT[32 * 128];   // [t][d]
    __shared__ unsigned short vT[128 * 32];   // [d][t]
    __shared__ unsigned short pW[8][16 * 32]; // per-wave P tile

    const int tid  = threadIdx.x;
    const int lane = tid & 31;
    const int w    = tid >> 5;
    const int half = lane >> 4;
    const int r    = lane & 15;

    const int q0 = blockIdx.x * 128 + w * 16;
    const int bh = blockIdx.y;
    const int b  = bh / NH;
    const int h  = bh % NH;

    const unsigned short* qbase = qws  + (size_t)bh * SEQ * HD;
    const unsigned short* kbase = kws  + (size_t)bh * SEQ * HD;
    const unsigned short* vbase = vtws + (size_t)bh * HD * SEQ;

    // resident Q fragments (hd chunks of 32)
    v16bf qf[4];
    #pragma unroll
    for (int c = 0; c < 4; c++)
        qf[c] = load_afrag(qbase + (size_t)(q0 + r) * HD + c * 32, half);

    v8f of[8];
    #pragma unroll
    for (int d = 0; d < 8; d++) of[d] = zero8();
    float mrow[8], lrow[8];
    #pragma unroll
    for (int vr = 0; vr < 8; vr++) { mrow[vr] = -1e30f; lrow[vr] = 0.0f; }

    const int qmax = blockIdx.x * 128 + 127;
    for (int t0 = 0; t0 <= qmax; t0 += 32) {
        // TDM: K tile [32 keys][128 hd] and V^T tile [128 hd][32 keys]
        if (w == 0) {
            tdm_load_2d(lds_off(kT), kbase + (size_t)t0 * HD, /*tile0=*/128,
                        /*tile1=*/32, /*stride0=*/HD);
            tdm_load_2d(lds_off(vT), vbase + t0, /*tile0=*/32,
                        /*tile1=*/128, /*stride0=*/SEQ);
            __builtin_amdgcn_s_wait_tensorcnt(0);
        }
        __syncthreads();

        // scores: 16 q rows x 32 keys
        v8f s0 = zero8(), s1 = zero8();
        #pragma unroll
        for (int c = 0; c < 4; c++) {
            s0 = wmma_bf16(qf[c], load_bfrag(&kT[(0  + r) * 128 + c * 32], half), s0);
            s1 = wmma_bf16(qf[c], load_bfrag(&kT[(16 + r) * 128 + c * 32], half), s1);
        }

        // causal mask + online softmax (row = vgpr index + 8*half, across 16 lanes)
        const int key0 = t0 + r;
        const int key1 = t0 + 16 + r;
        unsigned short* pbuf = pW[w];
        #pragma unroll
        for (int vr = 0; vr < 8; vr++) {
            const int qrow = q0 + half * 8 + vr;
            float a0 = (key0 > qrow) ? -10000.0f : s0[vr];
            float a1 = (key1 > qrow) ? -10000.0f : s1[vr];
            float mx = fmaxf(a0, a1);
            #pragma unroll
            for (int off = 1; off < 16; off <<= 1) mx = fmaxf(mx, __shfl_xor(mx, off, 32));
            float mnew  = fmaxf(mrow[vr], mx);
            float alpha = __expf(mrow[vr] - mnew);
            float p0 = __expf(a0 - mnew);
            float p1 = __expf(a1 - mnew);
            float sum = p0 + p1;
            #pragma unroll
            for (int off = 1; off < 16; off <<= 1) sum += __shfl_xor(sum, off, 32);
            lrow[vr] = lrow[vr] * alpha + sum;
            mrow[vr] = mnew;
            #pragma unroll
            for (int d = 0; d < 8; d++) of[d][vr] = of[d][vr] * alpha;
            pbuf[(half * 8 + vr) * 32 + r]      = f2bf(p0);
            pbuf[(half * 8 + vr) * 32 + 16 + r] = f2bf(p1);
        }

        // P (16x32) as A fragment, then PV accumulate over 8 hd chunks
        v16bf pf = load_afrag(&pbuf[r * 32], half);
        #pragma unroll
        for (int d = 0; d < 8; d++)
            of[d] = wmma_bf16(pf, load_bfrag(&vT[(d * 16 + r) * 32], half), of[d]);
        __syncthreads();
    }

    // normalize + store context bf16 [s*BSZ+b][h*HD+d]
    #pragma unroll
    for (int vr = 0; vr < 8; vr++) {
        const int qrow = q0 + half * 8 + vr;
        const float inv = 1.0f / lrow[vr];
        const size_t base = ((size_t)qrow * BSZ + b) * HID + h * HD;
        #pragma unroll
        for (int d = 0; d < 8; d++)
            ctx[base + d * 16 + r] = f2bf(of[d][vr] * inv);
    }
}

// ---------------------------------------------------------------------------
// Kernel 3: output projection.  out[m,n] = ctx[m,:]·Wd[n,:]  (fp32 out, no bias)
// ctx tile (bf16, no conversion) staged by TDM; Wd tile converted on VALU.
// ---------------------------------------------------------------------------
__global__ __launch_bounds__(256) void out_gemm_kernel(
    const unsigned short* __restrict__ ctx, const float* __restrict__ Wd,
    float* __restrict__ out)
{
    __shared__ unsigned short lA[128 * 32];
    __shared__ unsigned short lB[128 * 32];

    const int tid  = threadIdx.x;
    const int lane = tid & 31;
    const int w    = tid >> 5;
    const int half = lane >> 4;
    const int r    = lane & 15;

    const int m0 = blockIdx.y * 128;
    const int n0 = blockIdx.x * 128;
    const int wm = (w & 3) * 32;
    const int wn = (w >> 2) * 64;

    v8f acc[2][4];
    #pragma unroll
    for (int i = 0; i < 2; i++)
        #pragma unroll
        for (int j = 0; j < 4; j++) acc[i][j] = zero8();

    const int rowS = tid >> 1;
    const int colS = (tid & 1) * 16;

    for (int k0 = 0; k0 < HID; k0 += 32) {
        // TDM: ctx tile [128 rows][32 k] (bf16 copy)
        if (w == 0)
            tdm_load_2d(lds_off(lA), ctx + (size_t)m0 * HID + k0, /*tile0=*/32,
                        /*tile1=*/128, /*stride0=*/HID);
        {   // stage Wd tile (fp32 -> bf16)
            const float4* s4 = (const float4*)(Wd + (size_t)(n0 + rowS) * HID + k0 + colS);
            float4 f0 = s4[0], f1 = s4[1], f2 = s4[2], f3 = s4[3];
            uint4 lo, hi;
            lo.x = pk2(f0.x, f0.y); lo.y = pk2(f0.z, f0.w);
            lo.z = pk2(f1.x, f1.y); lo.w = pk2(f1.z, f1.w);
            hi.x = pk2(f2.x, f2.y); hi.y = pk2(f2.z, f2.w);
            hi.z = pk2(f3.x, f3.y); hi.w = pk2(f3.z, f3.w);
            uint4* d4 = (uint4*)&lB[rowS * 32 + colS];
            d4[0] = lo; d4[1] = hi;
        }
        if (w == 0) __builtin_amdgcn_s_wait_tensorcnt(0);
        __syncthreads();

        v16bf af[2], bfv[4];
        #pragma unroll
        for (int i = 0; i < 2; i++)
            af[i] = load_afrag(&lA[(wm + i * 16 + r) * 32], half);
        #pragma unroll
        for (int j = 0; j < 4; j++)
            bfv[j] = load_bfrag(&lB[(wn + j * 16 + r) * 32], half);
        #pragma unroll
        for (int i = 0; i < 2; i++)
            #pragma unroll
            for (int j = 0; j < 4; j++)
                acc[i][j] = wmma_bf16(af[i], bfv[j], acc[i][j]);
        __syncthreads();
    }

    #pragma unroll
    for (int i = 0; i < 2; i++)
        #pragma unroll
        for (int j = 0; j < 4; j++)
            #pragma unroll
            for (int vr = 0; vr < 8; vr++) {
                const int m = m0 + wm + i * 16 + half * 8 + vr;
                const int n = n0 + wn + j * 16 + r;
                out[(size_t)m * HID + n] = acc[i][j][vr];
            }
}

__global__ void bias_copy_kernel(const float* __restrict__ bias, float* __restrict__ out) {
    const int i = blockIdx.x * 256 + threadIdx.x;
    if (i < HID) out[(size_t)SEQ * BSZ * HID + i] = bias[i];
}

// ---------------------------------------------------------------------------
extern "C" void kernel_launch(void* const* d_in, const int* in_sizes, int n_in,
                              void* d_out, int out_size, void* d_ws, size_t ws_size,
                              hipStream_t stream) {
    (void)in_sizes; (void)n_in; (void)out_size; (void)ws_size;
    const float* hidden  = (const float*)d_in[0];
    // d_in[1] = attention_mask (causal, computed analytically) — unused
    const float* qkv_w   = (const float*)d_in[2];
    const float* qkv_b   = (const float*)d_in[3];
    const float* dense_w = (const float*)d_in[4];
    const float* dense_b = (const float*)d_in[5];
    float* out = (float*)d_out;

    const size_t qkvElems = (size_t)BSZ * NH * SEQ * HD;   // 8M bf16 each
    unsigned short* qws  = (unsigned short*)d_ws;
    unsigned short* kws  = qws + qkvElems;
    unsigned short* vtws = kws + qkvElems;
    unsigned short* ctx  = vtws + qkvElems;

    qkv_gemm_kernel<<<dim3(48, 32), 256, 0, stream>>>(hidden, qkv_w, qkv_b, qws, kws, vtws);
    attn_kernel<<<dim3(SEQ / 128, BSZ * NH), 256, 0, stream>>>(qws, kws, vtws, ctx);
    out_gemm_kernel<<<dim3(16, 32), 256, 0, stream>>>(ctx, dense_w, out);
    bias_copy_kernel<<<dim3(8), 256, 0, stream>>>(dense_b, out);
}
